// ConfigurationRelationDecoder_55018531061885
// MI455X (gfx1250) — matrix operand order repaired and checked by reading, not compiled, not run on hardware
//
#include <hip/hip_runtime.h>
#include <hip/hip_bf16.h>

typedef __attribute__((ext_vector_type(16))) _Float16 v16h;
typedef __attribute__((ext_vector_type(2)))  __fp16   v2fp16;
typedef __attribute__((ext_vector_type(8)))  float    v8f;

#define EPSF 1e-8f
#define NEGINF (-3.402823466e38f)

__device__ __forceinline__ unsigned int pack_f16x2(float a, float b) {
  union { v2fp16 h; unsigned int u; } p;
  p.h = __builtin_amdgcn_cvt_pkrtz(a, b);
  return p.u;
}

// ---------------------------------------------------------------------------
// Fragment loaders from LDS holding packed f16 pairs (1 dword = 2 halves,
// k-major within a row). ldPairs = dwords per row.
//
// A 16x32 f16 layout (ISA 7.12.2): lane L holds row L%16;
//   VGPR v: k = ((L&16)?8:0) + ((v&4)?16:0) + (v&3)*2
//   -> two contiguous 4-dword runs: base, base+8
// B 32x16 f16 layout (same k order as A): lane L holds col L%16;
//   VGPR v: k = ((L&16)?8:0) + ((v&4)?16:0) + (v&3)*2
//   -> two contiguous 4-dword runs: base, base+8
// ---------------------------------------------------------------------------
__device__ __forceinline__ v16h lds_a_frag(const unsigned int* __restrict__ pairs,
                                           int row, int ldPairs, int kstep, int lane) {
  int base = row * ldPairs + (kstep >> 1) + ((lane & 16) ? 4 : 0);
  union { v16h h; uint4 u[2]; } f;
  f.u[0] = *(const uint4*)(pairs + base);
  f.u[1] = *(const uint4*)(pairs + base + 8);
  return f.h;
}

__device__ __forceinline__ v16h lds_b_frag(const unsigned int* __restrict__ pairs,
                                           int col, int ldPairs, int kstep, int lane) {
  int base = col * ldPairs + (kstep >> 1) + ((lane & 16) ? 4 : 0);
  union { v16h h; uint4 u[2]; } f;
  f.u[0] = *(const uint4*)(pairs + base);
  f.u[1] = *(const uint4*)(pairs + base + 8);
  return f.h;
}

// ---------------------------------------------------------------------------
// Fused cosine-similarity + argmax over O=36 objects, norms computed in-block.
// obj: [B,I,36,300]  land: [B,128,300]  outIdx: [B,I,128]
// One block (256 thr) per (b,i); wave w owns landmark columns [16w,16w+16).
// A(48x320) and B(128x320) staged in LDS as packed f16 pairs.
// Dynamic LDS: (48*160 + 128*160)*4 + 176*4 = 113344 bytes.
// ---------------------------------------------------------------------------
__global__ void cos_argmax_wmma(const float* __restrict__ obj,
                                const float* __restrict__ land,
                                int* __restrict__ outIdx, int I) {
  extern __shared__ unsigned int smem[];
  unsigned int* sA = smem;              // 48 rows * 160 pair-dwords
  unsigned int* sB = smem + 48 * 160;   // 128 rows * 160 pair-dwords
  float* sN = (float*)(smem + 48 * 160 + 128 * 160); // 164 norms (A rows, B rows)
  const _Float16* sAh = (const _Float16*)sA;
  const _Float16* sBh = (const _Float16*)sB;

  int bi = blockIdx.x;
  int b = bi / I;
  int w = threadIdx.x >> 5, lane = threadIdx.x & 31;
  const float2* Ab2 = (const float2*)(obj + (long)bi * 36 * 300);
  const float2* Bb2 = (const float2*)(land + (long)b * 128 * 300);

  // ---- stage A (36x150 pairs -> 48x160, zero padded) ----
  for (int p = threadIdx.x; p < 36 * 150; p += 256) {
    int r = p / 150, kp = p - r * 150;
    float2 v = Ab2[r * 150 + kp];
    sA[r * 160 + kp] = pack_f16x2(v.x, v.y);
  }
  for (int p = threadIdx.x; p < 36 * 10; p += 256) {      // pad pair cols 150..159
    int r = p / 10, kp = 150 + (p - r * 10);
    sA[r * 160 + kp] = 0u;
  }
  for (int p = threadIdx.x; p < 12 * 160; p += 256)       // pad rows 36..47
    sA[36 * 160 + p] = 0u;
  // ---- stage B (128x150 pairs -> 128x160, zero padded) ----
  for (int p = threadIdx.x; p < 128 * 150; p += 256) {
    int r = p / 150, kp = p - r * 150;
    float2 v = Bb2[r * 150 + kp];
    sB[r * 160 + kp] = pack_f16x2(v.x, v.y);
  }
  for (int p = threadIdx.x; p < 128 * 10; p += 256) {
    int r = p / 10, kp = 150 + (p - r * 10);
    sB[r * 160 + kp] = 0u;
  }
  __syncthreads();

  // ---- row norms from staged LDS data (wave per row) ----
  for (int row = w; row < 36 + 128; row += 8) {
    const _Float16* src = (row < 36) ? (sAh + row * 320) : (sBh + (row - 36) * 320);
    float s = 0.f;
    for (int k = lane; k < 300; k += 32) { float v = (float)src[k]; s += v * v; }
    for (int off = 16; off > 0; off >>= 1) s += __shfl_xor(s, off, 32);
    if (lane == 0) sN[row] = sqrtf(s);
  }
  __syncthreads();

  int colBase = w * 16;
  int mycol = colBase + (lane & 15);
  int rowHalf = lane & 15;

  v8f acc[3] = {};
#pragma unroll 1
  for (int kstep = 0; kstep < 320; kstep += 32) {
    v16h bf = lds_b_frag(sB, mycol, 160, kstep, lane);
#pragma unroll
    for (int t = 0; t < 3; ++t) {
      v16h af = lds_a_frag(sA, t * 16 + rowHalf, 160, kstep, lane);
      acc[t] = __builtin_amdgcn_wmma_f32_16x16x32_f16(
          false, af, false, bf, (short)0, acc[t], false, false);
    }
  }

  // normalize (hw reciprocal) + per-lane argmax (ascending row order)
  float nb = sN[36 + mycol];
  float best = NEGINF; int bestIdx = 0;
  int rofs = (lane & 16) ? 8 : 0;
#pragma unroll
  for (int t = 0; t < 3; ++t) {
#pragma unroll
    for (int j = 0; j < 8; ++j) {
      int r = t * 16 + rofs + j;
      if (r < 36) {
        float den = sN[r] * nb;
        den = den > EPSF ? den : EPSF;
        float v = acc[t][j] * __builtin_amdgcn_rcpf(den);
        if (v > best) { best = v; bestIdx = r; }
      }
    }
  }
  float ob = __shfl_xor(best, 16, 32);
  int   oi = __shfl_xor(bestIdx, 16, 32);
  if (ob > best || (ob == best && oi < bestIdx)) { best = ob; bestIdx = oi; }
  if (lane < 16) outIdx[(long)bi * 128 + mycol] = bestIdx;
}

// ---------------------------------------------------------------------------
// WMMA GEMM: C[M,N] (+)= A[M,K] @ B[K,N].  A row-major, B [K,N] (k-major).
// M multiple of 16 (always 64 here), K even.  Block = 256 threads computes a
// 16 x 128 strip of C: wave w owns the 16-col subtile w.  Chunked LDS staging
// with packed f16 converts; branchless tails (clamped address + select).
// ---------------------------------------------------------------------------
__global__ void gemm_wmma(const float* __restrict__ A,
                          const float* __restrict__ B,
                          float* __restrict__ C,
                          int M, int N, int K, int accumulate) {
  __shared__ unsigned int sA[16 * 16];   // [row][kpair]
  __shared__ unsigned int sB[128 * 16];  // [col][kpair]
  int w = threadIdx.x >> 5, lane = threadIdx.x & 31;
  int rowBase = blockIdx.y * 16;
  int colBase = blockIdx.x * 128;

  v8f acc = {};
  int myc = colBase + w * 16 + (lane & 15);
  if (accumulate) {
#pragma unroll
    for (int j = 0; j < 8; ++j) {
      int r = rowBase + j + ((lane & 16) ? 8 : 0);
      if (myc < N) acc[j] = C[(long)r * N + myc];
    }
  }

#pragma unroll 1
  for (int kc = 0; kc < K; kc += 32) {
    __syncthreads();
    // stage A chunk: 16 rows x 16 pairs (one pair per thread)
    {
      int p = threadIdx.x;                        // 256 = 16*16
      int r = p >> 4, kp = p & 15;
      int gk0 = kc + kp * 2;
      int gp = gk0 < K ? gk0 : K - 2;             // K even -> pair stays in range
      const float2* A2 = (const float2*)A;
      float2 v = A2[((long)(rowBase + r) * K + gp) >> 1];
      bool ok = gk0 < K;
      sA[r * 16 + kp] = pack_f16x2(ok ? v.x : 0.f, ok ? v.y : 0.f);
    }
    // stage B chunk: 128 cols x 16 pairs (coalesced over cols)
    {
      int p = threadIdx.x;                        // 8 iters
#pragma unroll
      for (int it = 0; it < 8; ++it, p += 256) {
        int c = p & 127, kp = p >> 7;
        int gk0 = kc + kp * 2;
        int gk0c = gk0 < K ? gk0 : K - 1;
        int gk1c = gk0 + 1 < K ? gk0 + 1 : K - 1;
        int gc = colBase + c;
        int gcc = gc < N ? gc : N - 1;
        float v0 = B[(long)gk0c * N + gcc];
        float v1 = B[(long)gk1c * N + gcc];
        sB[c * 16 + kp] = pack_f16x2(gk0 < K ? v0 : 0.f, gk0 + 1 < K ? v1 : 0.f);
      }
    }
    if (kc + 32 < K) {
      __builtin_prefetch(&B[(long)(kc + 32) * N + colBase + (threadIdx.x & 127)], 0, 1);
      __builtin_prefetch(&A[(long)(rowBase + (threadIdx.x & 15)) * K + kc + 32], 0, 1);
    }
    __syncthreads();
    v16h a = lds_a_frag(sA, lane & 15, 16, 0, lane);
    v16h bfr = lds_b_frag(sB, w * 16 + (lane & 15), 16, 0, lane);
    acc = __builtin_amdgcn_wmma_f32_16x16x32_f16(
        false, a, false, bfr, (short)0, acc, false, false);
  }

#pragma unroll
  for (int j = 0; j < 8; ++j) {
    int r = rowBase + j + ((lane & 16) ? 8 : 0);
    if (myc < N) C[(long)r * N + myc] = acc[j];
  }
}

// ---------------------------------------------------------------------------
// Top-3 of weight[b, m/8] * mask[b, m] over m in [0,128). One thread per b.
// ---------------------------------------------------------------------------
__global__ void topk3_kernel(const float* __restrict__ wgt,   // [B,16]
                             const float* __restrict__ mask,  // [B,128]
                             int* __restrict__ out) {         // [B,3]
  int b = blockIdx.x * blockDim.x + threadIdx.x;
  if (b >= 64) return;
  float v0 = NEGINF, v1 = NEGINF, v2 = NEGINF;
  int i0 = 0, i1 = 0, i2 = 0;
  for (int m = 0; m < 128; ++m) {
    float v = wgt[b * 16 + (m >> 3)] * mask[b * 128 + m];
    if (v > v0)      { v2 = v1; i2 = i1; v1 = v0; i1 = i0; v0 = v; i0 = m; }
    else if (v > v1) { v2 = v1; i2 = i1; v1 = v; i1 = m; }
    else if (v > v2) { v2 = v; i2 = m; }
  }
  out[b * 3 + 0] = i0; out[b * 3 + 1] = i1; out[b * 3 + 2] = i2;
}

// ---------------------------------------------------------------------------
__global__ void gather_pano_kernel(const float* __restrict__ pano_obj, // [B,36,36,300]
                                   const int* __restrict__ pano_idx,   // [B,36,128]
                                   const int* __restrict__ top_idx,    // [B,3]
                                   float* __restrict__ out) {          // [B,36,900]
  int bi = blockIdx.x;
  int b = bi / 36;
  for (int e = threadIdx.x; e < 900; e += blockDim.x) {
    int j = e / 300, d = e % 300;
    int m = top_idx[b * 3 + j];
    int sel = pano_idx[(long)bi * 128 + m];
    out[(long)bi * 900 + e] = pano_obj[((long)bi * 36 + sel) * 300 + d];
  }
}

// ---------------------------------------------------------------------------
__global__ void action_embed_kernel(const float* __restrict__ action, // [B,128]
                                    const float* __restrict__ embW,   // [128,64]
                                    const float* __restrict__ embB,   // [64]
                                    float* __restrict__ x) {          // [B,3140]
  int t = blockIdx.x * blockDim.x + threadIdx.x;
  if (t >= 64 * 64) return;
  int b = t / 64, e = t % 64;
  float s = embB[e];
  for (int k = 0; k < 128; ++k) s += action[b * 128 + k] * embW[k * 64 + e];
  x[b * 3140 + e] = tanhf(s);
}

// ---------------------------------------------------------------------------
__global__ void feat_attn_kernel(const float* __restrict__ feature, // [B,36,2176]
                                 const float* __restrict__ pso,     // [B,36,900]
                                 const float* __restrict__ q1,      // [B,3076]
                                 float* __restrict__ x) {           // [B,3140]
  __shared__ float sA[36];
  int b = blockIdx.x;
  int w = threadIdx.x >> 5, lane = threadIdx.x & 31;
  const float* q = q1 + (long)b * 3076;
  for (int s = w; s < 36; s += 8) {
    float p = 0.f;
    const float* fr = feature + ((long)b * 36 + s) * 2176;
    for (int d = lane; d < 2176; d += 32) p += fr[d] * q[d];
    const float* pr = pso + ((long)b * 36 + s) * 900;
    for (int d = lane; d < 900; d += 32) p += pr[d] * q[2176 + d];
    for (int off = 16; off > 0; off >>= 1) p += __shfl_xor(p, off, 32);
    if (lane == 0) sA[s] = p;
  }
  __syncthreads();
  if (threadIdx.x == 0) {
    float mx = sA[0];
    for (int s = 1; s < 36; ++s) mx = fmaxf(mx, sA[s]);
    float sum = 0.f;
    for (int s = 0; s < 36; ++s) { float e = expf(sA[s] - mx); sA[s] = e; sum += e; }
    float inv = 1.f / sum;
    for (int s = 0; s < 36; ++s) sA[s] *= inv;
  }
  __syncthreads();
  for (int d = threadIdx.x; d < 3076; d += blockDim.x) {
    float s2 = 0.f;
    if (d < 2176)
      for (int s = 0; s < 36; ++s) s2 += sA[s] * feature[((long)b * 36 + s) * 2176 + d];
    else {
      int dd = d - 2176;
      for (int s = 0; s < 36; ++s) s2 += sA[s] * pso[((long)b * 36 + s) * 900 + dd];
    }
    x[b * 3140 + 64 + d] = s2;
  }
}

__global__ void bias_init_kernel(const float* __restrict__ bias,
                                 float* __restrict__ gates) {
  int t = blockIdx.x * blockDim.x + threadIdx.x;
  if (t < 64 * 2048) gates[t] = bias[t % 2048];
}

__global__ void lstm_kernel(const float* __restrict__ gates, // [B,2048]
                            const float* __restrict__ c0,    // [B,512]
                            float* __restrict__ h1_out,
                            float* __restrict__ c1_out) {
  int t = blockIdx.x * blockDim.x + threadIdx.x;
  if (t >= 64 * 512) return;
  int b = t / 512, h = t % 512;
  const float* g = gates + (long)b * 2048;
  float ig = 1.f / (1.f + expf(-g[h]));
  float fg = 1.f / (1.f + expf(-g[512 + h]));
  float gg = tanhf(g[1024 + h]);
  float og = 1.f / (1.f + expf(-g[1536 + h]));
  float c1 = fg * c0[t] + ig * gg;
  c1_out[t] = c1;
  h1_out[t] = og * tanhf(c1);
}

__global__ void ctx_attn_kernel(const float* __restrict__ ctx,   // [B,16,512]
                                const float* __restrict__ q2,    // [B,512]
                                const unsigned char* __restrict__ cmask, // [B,16]
                                const float* __restrict__ h1,    // [B,512]
                                float* __restrict__ attn_out,    // [B,16]
                                float* __restrict__ cat2) {      // [B,1024]
  __shared__ float sA[16];
  int b = blockIdx.x;
  int w = threadIdx.x >> 5, lane = threadIdx.x & 31;
  for (int s = w; s < 16; s += 8) {
    float p = 0.f;
    const float* cr = ctx + ((long)b * 16 + s) * 512;
    for (int d = lane; d < 512; d += 32) p += cr[d] * q2[(long)b * 512 + d];
    for (int off = 16; off > 0; off >>= 1) p += __shfl_xor(p, off, 32);
    if (lane == 0) sA[s] = cmask[b * 16 + s] ? NEGINF : p;
  }
  __syncthreads();
  if (threadIdx.x == 0) {
    float mx = sA[0];
    for (int s = 1; s < 16; ++s) mx = fmaxf(mx, sA[s]);
    float sum = 0.f;
    for (int s = 0; s < 16; ++s) { float e = expf(sA[s] - mx); sA[s] = e; sum += e; }
    float inv = 1.f / sum;
    for (int s = 0; s < 16; ++s) sA[s] *= inv;
  }
  __syncthreads();
  if (threadIdx.x < 16) attn_out[b * 16 + threadIdx.x] = sA[threadIdx.x];
  for (int d = threadIdx.x; d < 512; d += blockDim.x) {
    float s2 = 0.f;
    for (int s = 0; s < 16; ++s) s2 += sA[s] * ctx[((long)b * 16 + s) * 512 + d];
    cat2[b * 1024 + d] = s2;
    cat2[b * 1024 + 512 + d] = h1[b * 512 + d];
  }
}

__global__ void tanh_kernel(const float* __restrict__ in, float* __restrict__ out, int n) {
  int t = blockIdx.x * blockDim.x + threadIdx.x;
  if (t < n) out[t] = tanhf(in[t]);
}

__global__ void cand_extra_kernel(const float* __restrict__ cand_obj, // [B,32,36,300]
                                  const int* __restrict__ cand_idx,   // [B,32,128]
                                  const int* __restrict__ ctop,       // [B,3]
                                  const float* __restrict__ lrel,     // [B,128,6]
                                  const float* __restrict__ lrelmask, // [B,128]
                                  const float* __restrict__ crel,     // [B,32,6]
                                  float* __restrict__ out) {          // [B,32,921]
  int bi = blockIdx.x;
  int b = bi / 32;
  for (int e = threadIdx.x; e < 921; e += blockDim.x) {
    int j = e / 307, r = e % 307;
    int m = ctop[b * 3 + j];
    float val;
    if (r < 300) {
      int sel = cand_idx[(long)bi * 128 + m];
      val = cand_obj[((long)bi * 36 + sel) * 300 + r];
    } else if (r < 306) {
      val = crel[bi * 6 + (r - 300)] * lrelmask[b * 128 + m];
    } else {
      float s = 0.f;
      for (int k = 0; k < 6; ++k) s += crel[bi * 6 + k] * lrel[(b * 128 + m) * 6 + k];
      val = s;
    }
    out[(long)bi * 921 + e] = val;
  }
}

__global__ void logit_kernel(const float* __restrict__ cand_feat,  // [B,32,2176]
                             const float* __restrict__ cand_extra, // [B,32,921]
                             const float* __restrict__ q3,         // [B,3097]
                             float* __restrict__ logit) {          // [B,32]
  int b = blockIdx.x;
  int w = threadIdx.x >> 5, lane = threadIdx.x & 31;
  const float* q = q3 + (long)b * 3097;
  for (int s = w; s < 32; s += 8) {
    float p = 0.f;
    const float* cr = cand_feat + ((long)b * 32 + s) * 2176;
    for (int d = lane; d < 2176; d += 32) p += cr[d] * q[d];
    const float* er = cand_extra + ((long)b * 32 + s) * 921;
    for (int d = lane; d < 921; d += 32) p += er[d] * q[2176 + d];
    for (int off = 16; off > 0; off >>= 1) p += __shfl_xor(p, off, 32);
    if (lane == 0) logit[b * 32 + s] = p;
  }
}

// ---------------------------------------------------------------------------
extern "C" void kernel_launch(void* const* d_in, const int* in_sizes, int n_in,
                              void* d_out, int out_size, void* d_ws, size_t ws_size,
                              hipStream_t stream) {
  const float* action    = (const float*)d_in[0];
  const float* feature   = (const float*)d_in[1];
  const float* cand_feat = (const float*)d_in[2];
  const float* prev_h1   = (const float*)d_in[3];
  const float* c_0       = (const float*)d_in[4];
  const float* ctx       = (const float*)d_in[5];
  const float* s_0       = (const float*)d_in[6];
  const float* land      = (const float*)d_in[7];   // [B,128,300]
  const float* cand_obj  = (const float*)d_in[8];   // [B,32,36,300]
  const float* lmask     = (const float*)d_in[9];   // [B,128]
  const float* lrel      = (const float*)d_in[10];  // [B,128,6]
  const float* lrelmask  = (const float*)d_in[11];  // [B,128]
  const float* crel      = (const float*)d_in[12];  // [B,32,6]
  const float* pano_obj  = (const float*)d_in[13];  // [B,36,36,300]
  const float* embW      = (const float*)d_in[14];
  const float* embB      = (const float*)d_in[15];
  const float* W_ih      = (const float*)d_in[16];  // [3140,2048]
  const float* W_hh      = (const float*)d_in[17];  // [512,2048]
  const float* b_lstm    = (const float*)d_in[18];
  const float* feat_in_W = (const float*)d_in[19];  // [512,3076]
  const float* att_in_W  = (const float*)d_in[20];  // [512,512]
  const float* att_out_W = (const float*)d_in[21];  // [1024,512]
  const float* cand_in_W = (const float*)d_in[22];  // [512,3097]
  const unsigned char* ctx_mask = (const unsigned char*)d_in[23];

  float* out      = (float*)d_out;
  float* h1_out   = out;           // [64,512]
  float* c1_out   = out + 32768;   // [64,512]
  float* logit_o  = out + 65536;   // [64,32]
  float* ht_out   = out + 67584;   // [64,512]
  float* attn_out = out + 100352;  // [64,16]

  float* ws = (float*)d_ws;
  float* pso      = ws;                 // 2073600  [B,36,900]
  float* q1       = ws + 2073600;       // 196864   [B,3076]
  float* x        = ws + 2270464;       // 200960   [B,3140]
  float* gates    = ws + 2471424;       // 131072   [B,2048]
  float* q2       = ws + 2602496;       // 32768
  float* cat2     = ws + 2635264;       // 65536    [B,1024]
  float* htpre    = ws + 2700800;       // 32768
  float* q3       = ws + 2733568;       // 198208   [B,3097]
  float* cextra   = ws + 2931776;       // 1886208  [B,32,921]
  int* wi       = (int*)(ws + 4817984);
  int* pano_idx = wi;                   // 294912   [B,36,128]
  int* cand_idx = wi + 294912;          // 262144   [B,32,128]
  int* top_idx  = wi + 557056;          // 192
  int* ctop     = wi + 557248;          // 192

  const size_t cosLds = (48 * 160 + 128 * 160) * 4 + 176 * 4; // 113344 B

  // --- cosine argmax (WMMA, norms fused) ---
  cos_argmax_wmma<<<64 * 36, 256, cosLds, stream>>>(pano_obj, land, pano_idx, 36);
  cos_argmax_wmma<<<64 * 32, 256, cosLds, stream>>>(cand_obj, land, cand_idx, 32);

  // --- pano top-3 & gather ---
  topk3_kernel<<<1, 64, 0, stream>>>(s_0, lmask, top_idx);
  gather_pano_kernel<<<64 * 36, 256, 0, stream>>>(pano_obj, pano_idx, top_idx, pso);

  // --- q1 = prev_h1 @ feat_in_W (WMMA) ---
  gemm_wmma<<<dim3((3076 + 127) / 128, 4), 256, 0, stream>>>(prev_h1, feat_in_W, q1,
                                                             64, 3076, 512, 0);

  // --- action embed + feature attention -> x = [embed | attn_feat] ---
  action_embed_kernel<<<(64 * 64 + 255) / 256, 256, 0, stream>>>(action, embW, embB, x);
  feat_attn_kernel<<<64, 256, 0, stream>>>(feature, pso, q1, x);

  // --- LSTM gates = b + x@W_ih + h@W_hh (WMMA accumulate) ---
  bias_init_kernel<<<(64 * 2048 + 255) / 256, 256, 0, stream>>>(b_lstm, gates);
  gemm_wmma<<<dim3(2048 / 128, 4), 256, 0, stream>>>(x, W_ih, gates, 64, 2048, 3140, 1);
  gemm_wmma<<<dim3(2048 / 128, 4), 256, 0, stream>>>(prev_h1, W_hh, gates, 64, 2048, 512, 1);
  lstm_kernel<<<(64 * 512 + 255) / 256, 256, 0, stream>>>(gates, c_0, h1_out, c1_out);

  // --- ctx attention ---
  gemm_wmma<<<dim3(512 / 128, 4), 256, 0, stream>>>(h1_out, att_in_W, q2, 64, 512, 512, 0);
  ctx_attn_kernel<<<64, 256, 0, stream>>>(ctx, q2, ctx_mask, h1_out, attn_out, cat2);
  gemm_wmma<<<dim3(512 / 128, 4), 256, 0, stream>>>(cat2, att_out_W, htpre, 64, 512, 1024, 0);
  tanh_kernel<<<(64 * 512 + 255) / 256, 256, 0, stream>>>(htpre, ht_out, 64 * 512);

  // --- candidate side ---
  gemm_wmma<<<dim3((3097 + 127) / 128, 4), 256, 0, stream>>>(ht_out, cand_in_W, q3,
                                                             64, 3097, 512, 0);
  topk3_kernel<<<1, 64, 0, stream>>>(attn_out, lmask, ctop);
  cand_extra_kernel<<<64 * 32, 256, 0, stream>>>(cand_obj, cand_idx, ctop, lrel,
                                                 lrelmask, crel, cextra);
  logit_kernel<<<64, 256, 0, stream>>>(cand_feat, cextra, q3, logit_o);
}
